// Int8LlamaMLP_29025388986471
// MI455X (gfx1250) — 8-bit matrix operand order repaired and checked by reading, compiled, not run on hardware
//
#include <hip/hip_runtime.h>
#include <stdint.h>

typedef int v8i __attribute__((ext_vector_type(8)));
typedef int v4i __attribute__((ext_vector_type(4)));
typedef int v2i __attribute__((ext_vector_type(2)));

static constexpr int Hdim = 4096;
static constexpr int Fdim = 11008;
static constexpr int Mtot = 4096;

#if defined(__has_builtin)
#  if __has_builtin(__builtin_amdgcn_global_load_async_to_lds_b128)
#    define USE_ASYNC_BUILTIN 1
#  endif
#  if __has_builtin(__builtin_amdgcn_s_wait_asynccnt)
#    define USE_WAIT_BUILTIN 1
#  endif
#endif

typedef __attribute__((address_space(1))) v4i gv4i;
typedef __attribute__((address_space(3))) v4i lv4i;

__device__ __forceinline__ void async_ld16(const int8_t* g, const int8_t* l) {
#if USE_ASYNC_BUILTIN
  __builtin_amdgcn_global_load_async_to_lds_b128(
      (gv4i*)(uintptr_t)g, (lv4i*)(uintptr_t)(uint32_t)(uintptr_t)l, 0, 0);
#else
  const uint32_t loff = (uint32_t)(uintptr_t)l;
  asm volatile("global_load_async_to_lds_b128 %0, %1, off"
               :: "v"(loff), "v"((uint64_t)(uintptr_t)g)
               : "memory");
#endif
}

template <int N>
__device__ __forceinline__ void wait_async() {
#if USE_WAIT_BUILTIN
  __builtin_amdgcn_s_wait_asynccnt(N);
#else
  asm volatile("s_wait_asynccnt %0" :: "n"(N) : "memory");
#endif
}

__device__ __forceinline__ v8i wmma_iu8(v8i a, v8i b, v8i c) {
  return __builtin_amdgcn_wmma_i32_16x16x64_iu8(true, a, true, b, c, false, false);
}

__device__ __forceinline__ int quant_i8(float v) {
  v = rintf(v);
  v = fminf(fmaxf(v, -128.0f), 127.0f);
  return (int)v;
}

__global__ __launch_bounds__(256)
void int8_gateup_wmma(const int8_t* __restrict__ x,
                      const int8_t* __restrict__ Wg, const int8_t* __restrict__ bg,
                      const int8_t* __restrict__ Wu, const int8_t* __restrict__ bu,
                      const float* __restrict__ p_alpha_g, const float* __restrict__ p_beta_g,
                      const float* __restrict__ p_alpha_u, const float* __restrict__ p_beta_u,
                      int8_t* __restrict__ prod8)
{
  __shared__ int8_t sA [2][128 * 128] __attribute__((aligned(16)));
  __shared__ int8_t sBg[2][ 64 * 128] __attribute__((aligned(16)));
  __shared__ int8_t sBu[2][ 64 * 128] __attribute__((aligned(16)));

  const int t    = threadIdx.x;
  const int lane = t & 31;
  const int wid  = t >> 5;
  const int wm   = wid & 3;
  const int wn   = wid >> 2;
  const int l16  = lane & 15;
  const int lh   = lane >> 4;

  const int f0 = blockIdx.x * 64;
  const int m0 = blockIdx.y * 128;

  v8i accg[2][2], accu[2][2];
#pragma unroll
  for (int i = 0; i < 2; ++i)
#pragma unroll
    for (int j = 0; j < 2; ++j) { accg[i][j] = {}; accu[i][j] = {}; }

  const int arow  = t >> 1;
  const int acolb = (t & 1) * 64;
  const int brow  = t >> 2;
  const int bcolb = (t & 3) * 32;

  auto issue_stage = [&](int stage, int buf) {
    const int k0 = stage * 128;
    const int8_t* ga = x + (size_t)(m0 + arow) * Hdim + k0 + acolb;
    const int8_t* la = &sA[buf][arow * 128 + acolb];
#pragma unroll
    for (int c = 0; c < 4; ++c) async_ld16(ga + c * 16, la + c * 16);
    const int8_t* gg = Wg + (size_t)(f0 + brow) * Hdim + k0 + bcolb;
    const int8_t* gu = Wu + (size_t)(f0 + brow) * Hdim + k0 + bcolb;
    const int8_t* lg = &sBg[buf][brow * 128 + bcolb];
    const int8_t* lu = &sBu[buf][brow * 128 + bcolb];
#pragma unroll
    for (int c = 0; c < 2; ++c) {
      async_ld16(gg + c * 16, lg + c * 16);
      async_ld16(gu + c * 16, lu + c * 16);
    }
  };

  auto compute_stage = [&](int buf) {
#pragma unroll
    for (int s = 0; s < 2; ++s) {
      const int koff = s * 64;
      v8i afr[2];
#pragma unroll
      for (int mt = 0; mt < 2; ++mt) {
        const int8_t* base = &sA[buf][(wm * 32 + mt * 16 + l16) * 128 + koff + lh * 16];
        v8i a;
#pragma unroll
        for (int c = 0; c < 4; ++c) {
          v2i tt = *(const v2i*)(base + (c & 1) * 8 + (c >> 1) * 32);
          a[2 * c] = tt[0]; a[2 * c + 1] = tt[1];
        }
        afr[mt] = a;
      }
      v8i bgfr[2], bufr[2];
#pragma unroll
      for (int nt = 0; nt < 2; ++nt) {
        const int ro = (wn * 32 + nt * 16 + l16) * 128 + koff + lh * 16;
        v8i b, c;
#pragma unroll
        for (int p = 0; p < 2; ++p) {
          v4i tg = *(const v4i*)(&sBg[buf][ro + p * 32]);
          v4i tu = *(const v4i*)(&sBu[buf][ro + p * 32]);
#pragma unroll
          for (int i = 0; i < 4; ++i) { b[4 * p + i] = tg[i]; c[4 * p + i] = tu[i]; }
        }
        bgfr[nt] = b; bufr[nt] = c;
      }
#pragma unroll
      for (int mt = 0; mt < 2; ++mt)
#pragma unroll
        for (int nt = 0; nt < 2; ++nt) {
          accg[mt][nt] = wmma_iu8(afr[mt], bgfr[nt], accg[mt][nt]);
          accu[mt][nt] = wmma_iu8(afr[mt], bufr[nt], accu[mt][nt]);
        }
    }
  };

  constexpr int NIT = Hdim / 128;
  issue_stage(0, 0);
  for (int it = 0; it < NIT; ++it) {
    const int cur = it & 1;
    if (it + 1 < NIT) {
      issue_stage(it + 1, cur ^ 1);
      wait_async<8>();
    } else {
      wait_async<0>();
    }
    __syncthreads();
    compute_stage(cur);
    __syncthreads();
  }

  const float alpha_g = *p_alpha_g, beta_g = *p_beta_g;
  const float alpha_u = *p_alpha_u, beta_u = *p_beta_u;

#pragma unroll
  for (int nt = 0; nt < 2; ++nt) {
    const int f = f0 + wn * 32 + nt * 16 + l16;
    const float bgf = (float)bg[f];
    const float buf = (float)bu[f];
#pragma unroll
    for (int mt = 0; mt < 2; ++mt) {
#pragma unroll
      for (int r = 0; r < 8; ++r) {
        const int m = m0 + wm * 32 + mt * 16 + lh * 8 + r;
        float gf = alpha_g * (float)accg[mt][nt][r] + beta_g * bgf;
        gf = fmaxf(gf, 0.0f);
        const int gq = quant_i8(gf);
        const float uf = alpha_u * (float)accu[mt][nt][r] + beta_u * buf;
        const int uq = quant_i8(uf);
        prod8[(size_t)m * Fdim + f] = (int8_t)(gq * uq);
      }
    }
  }
}

__global__ __launch_bounds__(256)
void int8_down_wmma(const int8_t* __restrict__ p8,
                    const int8_t* __restrict__ Wd,
                    const float* __restrict__ p_alpha_d,
                    const float* __restrict__ bias_d,
                    float* __restrict__ out)
{
  __shared__ int8_t sA[2][128 * 128] __attribute__((aligned(16)));
  __shared__ int8_t sB[2][ 64 * 128] __attribute__((aligned(16)));

  const int t    = threadIdx.x;
  const int lane = t & 31;
  const int wid  = t >> 5;
  const int wm   = wid & 3;
  const int wn   = wid >> 2;
  const int l16  = lane & 15;
  const int lh   = lane >> 4;

  const int h0 = blockIdx.x * 64;
  const int m0 = blockIdx.y * 128;

  v8i acc[2][2];
#pragma unroll
  for (int i = 0; i < 2; ++i)
#pragma unroll
    for (int j = 0; j < 2; ++j) acc[i][j] = {};

  const int arow  = t >> 1;
  const int acolb = (t & 1) * 64;
  const int brow  = t >> 2;
  const int bcolb = (t & 3) * 32;

  auto issue_stage = [&](int stage, int buf) {
    const int k0 = stage * 128;
    const int8_t* ga = p8 + (size_t)(m0 + arow) * Fdim + k0 + acolb;
    const int8_t* la = &sA[buf][arow * 128 + acolb];
#pragma unroll
    for (int c = 0; c < 4; ++c) async_ld16(ga + c * 16, la + c * 16);
    const int8_t* gb = Wd + (size_t)(h0 + brow) * Fdim + k0 + bcolb;
    const int8_t* lb = &sB[buf][brow * 128 + bcolb];
#pragma unroll
    for (int c = 0; c < 2; ++c) async_ld16(gb + c * 16, lb + c * 16);
  };

  auto compute_stage = [&](int buf) {
#pragma unroll
    for (int s = 0; s < 2; ++s) {
      const int koff = s * 64;
      v8i afr[2];
#pragma unroll
      for (int mt = 0; mt < 2; ++mt) {
        const int8_t* base = &sA[buf][(wm * 32 + mt * 16 + l16) * 128 + koff + lh * 16];
        v8i a;
#pragma unroll
        for (int c = 0; c < 4; ++c) {
          v2i tt = *(const v2i*)(base + (c & 1) * 8 + (c >> 1) * 32);
          a[2 * c] = tt[0]; a[2 * c + 1] = tt[1];
        }
        afr[mt] = a;
      }
      v8i bfr[2];
#pragma unroll
      for (int nt = 0; nt < 2; ++nt) {
        const int ro = (wn * 32 + nt * 16 + l16) * 128 + koff + lh * 16;
        v8i b;
#pragma unroll
        for (int p = 0; p < 2; ++p) {
          v4i tb = *(const v4i*)(&sB[buf][ro + p * 32]);
#pragma unroll
          for (int i = 0; i < 4; ++i) b[4 * p + i] = tb[i];
        }
        bfr[nt] = b;
      }
#pragma unroll
      for (int mt = 0; mt < 2; ++mt)
#pragma unroll
        for (int nt = 0; nt < 2; ++nt)
          acc[mt][nt] = wmma_iu8(afr[mt], bfr[nt], acc[mt][nt]);
    }
  };

  constexpr int NIT = Fdim / 128;
  issue_stage(0, 0);
  for (int it = 0; it < NIT; ++it) {
    const int cur = it & 1;
    if (it + 1 < NIT) {
      issue_stage(it + 1, cur ^ 1);
      wait_async<6>();
    } else {
      wait_async<0>();
    }
    __syncthreads();
    compute_stage(cur);
    __syncthreads();
  }

  const float alpha_d = *p_alpha_d;
#pragma unroll
  for (int nt = 0; nt < 2; ++nt) {
    const int h = h0 + wn * 32 + nt * 16 + l16;
    const float bd = bias_d[h];
#pragma unroll
    for (int mt = 0; mt < 2; ++mt) {
#pragma unroll
      for (int r = 0; r < 8; ++r) {
        const int m = m0 + wm * 32 + mt * 16 + lh * 8 + r;
        out[(size_t)m * Hdim + h] = alpha_d * (float)acc[mt][nt][r] + bd;
      }
    }
  }
}

extern "C" void kernel_launch(void* const* d_in, const int* in_sizes, int n_in,
                              void* d_out, int out_size, void* d_ws, size_t ws_size,
                              hipStream_t stream) {
  const int8_t* x   = (const int8_t*)d_in[0];
  const int8_t* Wg  = (const int8_t*)d_in[1];
  const int8_t* bg  = (const int8_t*)d_in[2];
  const int8_t* Wu  = (const int8_t*)d_in[3];
  const int8_t* bu  = (const int8_t*)d_in[4];
  const int8_t* Wd  = (const int8_t*)d_in[5];
  const float* alpha_g = (const float*)d_in[6];
  const float* beta_g  = (const float*)d_in[7];
  const float* alpha_u = (const float*)d_in[8];
  const float* beta_u  = (const float*)d_in[9];
  const float* alpha_d = (const float*)d_in[10];
  const float* bias_d  = (const float*)d_in[11];

  int8_t* prod8 = (int8_t*)d_ws;
  float*  out   = (float*)d_out;

  dim3 blk(256, 1, 1);
  dim3 g1(Fdim / 64, Mtot / 128, 1);
  int8_gateup_wmma<<<g1, blk, 0, stream>>>(x, Wg, bg, Wu, bu,
                                           alpha_g, beta_g, alpha_u, beta_u, prod8);

  dim3 g2(Hdim / 64, Mtot / 128, 1);
  int8_down_wmma<<<g2, blk, 0, stream>>>(prod8, Wd, alpha_d, bias_d, out);
}
